// GRUCore_36627481101172
// MI455X (gfx1250) — matrix operand order repaired and checked by reading, not compiled, not run on hardware
//
#include <hip/hip_runtime.h>
#include <hip/hip_bf16.h>

// ---------------------------------------------------------------------------
// LayerNorm-GRU for MI455X (gfx1250, wave32, WMMA).
//
// Shapes: B=256, T=128, D=1024, H=1024, G=3*H=3072, K2=D+H=2048.
//
//  * Fused input+recurrent GEMM per step: gates = [x_t | h*mask] @ [Wih|Whh]^T
//    (no 384MB "gi" intermediate). bf16 hi/lo weights (24MB) stay L2-resident
//    across all 128 sequential steps (192MB L2).
//  * bf16x3 split precision (hi*hi + hi*lo + lo*hi) => ~fp32 accuracy on the
//    16x16x32 bf16 WMMA pipe.
//  * ALL fp32->bf16 conversion is hoisted out of the sequential inner loop:
//      - x is pre-converted once (parallel) into bf16 hi/lo, pre-swizzled in
//        the CDNA5 A-fragment lane order (K-index bit3<->bit4 swap), so each
//        lane's A fragment is one contiguous 32B load.
//      - each step emits h_new as swizzled bf16 hi/lo into a ping-pong buffer
//        consumed by the next step; the is_init reset is a branchless select
//        on the loaded fragment.
//    Inner loop = 16 b128 loads + 9 WMMA + trivial VALU.
//  * 128 stream-ordered launches give the device-wide sync the scan needs.
// ---------------------------------------------------------------------------

#define Bc 256
#define Tc 128
#define Dc 1024
#define Hc 1024
#define Gc 3072
#define K2 2048

typedef __attribute__((ext_vector_type(16))) __bf16 v16bf;
typedef __attribute__((ext_vector_type(8)))  float  v8f;

__device__ __forceinline__ float sigmoidf_(float v) {
    return 1.0f / (1.0f + __expf(-v));
}

// A-fragment swizzle: within each 32-wide K block the lane order is
// {0..7,16..23 | 8..15,24..31} == swap of bits 3 and 4 of the K index.
__device__ __forceinline__ int swz(int j) {
    return (j & ~0x18) | ((j & 0x08) << 1) | ((j & 0x10) >> 1);
}

// 3-term split-precision accumulate: c += Ah*Bh + Ah*Bl + Al*Bh
__device__ __forceinline__ v8f mm3(v16bf ah, v16bf al, v16bf bh, v16bf bl, v8f c) {
    c = __builtin_amdgcn_wmma_f32_16x16x32_bf16(false, ah, false, bh, (short)0, c, false, false);
    c = __builtin_amdgcn_wmma_f32_16x16x32_bf16(false, ah, false, bl, (short)0, c, false, false);
    c = __builtin_amdgcn_wmma_f32_16x16x32_bf16(false, al, false, bh, (short)0, c, false, false);
    return c;
}

// ---------------------------------------------------------------------------
// Kernel 1: concatenated bf16 hi/lo weights Wc[g][k] (k<1024: W_ih, else W_hh),
// row-major stride K2: B-fragment per lane is one contiguous 32B load.
// ---------------------------------------------------------------------------
__global__ __launch_bounds__(256) void conv_w(const float* __restrict__ W_ih,
                                              const float* __restrict__ W_hh,
                                              __bf16* __restrict__ Wh,
                                              __bf16* __restrict__ Wl) {
    long i = (long)blockIdx.x * blockDim.x + threadIdx.x;
    if (i >= (long)Gc * K2) return;
    int g = (int)(i >> 11);
    int k = (int)(i & (K2 - 1));
    float v = (k < Dc) ? W_ih[(long)g * Dc + k] : W_hh[(long)g * Hc + (k - Dc)];
    __bf16 h = (__bf16)v;
    Wh[i - k + swz(k)] = h;
    Wl[i - k + swz(k)] = (__bf16)(v - (float)h);
}

// ---------------------------------------------------------------------------
// Kernel 2: x -> swizzled bf16 hi/lo (row layout preserved, K bit3<->4 swap).
// ---------------------------------------------------------------------------
__global__ __launch_bounds__(256) void conv_x(const float* __restrict__ x,
                                              __bf16* __restrict__ xh,
                                              __bf16* __restrict__ xl) {
    long i = (long)blockIdx.x * blockDim.x + threadIdx.x;
    if (i >= (long)Bc * Tc * Dc) return;
    long rowbase = i & ~(long)(Dc - 1);
    int  j       = (int)(i & (Dc - 1));
    long o       = rowbase + swz(j);
    float v  = x[i];
    __bf16 h = (__bf16)v;
    xh[o] = h;
    xl[o] = (__bf16)(v - (float)h);
}

// ---------------------------------------------------------------------------
// Kernel 3: hx -> swizzled bf16 hi/lo ping-pong buffer 0 (for step t=0).
// ---------------------------------------------------------------------------
__global__ __launch_bounds__(256) void conv_h0(const float* __restrict__ hx,
                                               __bf16* __restrict__ hh,
                                               __bf16* __restrict__ hl) {
    long i = (long)blockIdx.x * blockDim.x + threadIdx.x;
    if (i >= (long)Bc * Hc) return;
    long rowbase = i & ~(long)(Hc - 1);
    int  j       = (int)(i & (Hc - 1));
    long o       = rowbase + swz(j);
    float v  = hx[i];
    __bf16 h = (__bf16)v;
    hh[o] = h;
    hl[o] = (__bf16)(v - (float)h);
}

// ---------------------------------------------------------------------------
// Kernel 4: one GRU time step. grid = (H/16, B/16), block = 1 wave (32).
// Each wave produces a 16(batch) x 16(hidden) tile of h_new, computing all
// three gate tiles so the recurrence update stays in-register.
// ---------------------------------------------------------------------------
__global__ __launch_bounds__(32) void gru_step(const __bf16* __restrict__ xh,    // (B*T, D) swizzled
                                               const __bf16* __restrict__ xl,
                                               const __bf16* __restrict__ hbr_h, // (B, H) swizzled, read
                                               const __bf16* __restrict__ hbr_l,
                                               __bf16* __restrict__ hbw_h,       // (B, H) swizzled, write
                                               __bf16* __restrict__ hbw_l,
                                               const float* __restrict__ hprev,  // fp32, row stride hstride
                                               long hstride,
                                               const unsigned char* __restrict__ is_init, // (B,T) bool
                                               const __bf16* __restrict__ Wh,
                                               const __bf16* __restrict__ Wl,
                                               const float* __restrict__ b_ih,
                                               const float* __restrict__ b_hh,
                                               float* __restrict__ hseq,         // (B,T,H)
                                               int t) {
    const int lane = threadIdx.x;
    const int hs   = lane >> 4;   // which K sub-chunk this lane covers
    const int lm   = lane & 15;   // row (A) / col (B,C) within tile
    const int nh   = blockIdx.x;  // hidden tile 0..63
    const int m0   = blockIdx.y * 16;

    // Per-lane A-row sources (row = lm of the tile).
    const int  arow = m0 + lm;
    const bool init = is_init[(long)arow * Tc + t] != 0;
    const __bf16* pxh = xh + ((long)arow * Tc + t) * Dc + hs * 16;
    const __bf16* pxl = xl + ((long)arow * Tc + t) * Dc + hs * 16;
    const __bf16* phh = hbr_h + (long)arow * Hc + hs * 16;
    const __bf16* phl = hbr_l + (long)arow * Hc + hs * 16;

    // Per-lane B rows: lane = column of B = row of Wc; K chunk offset hs*16.
    const long g_r = (long)(nh * 16 + lm);
    const __bf16* pWh_r = Wh + g_r * K2 + hs * 16;
    const __bf16* pWl_r = Wl + g_r * K2 + hs * 16;
    const __bf16* pWh_z = pWh_r + (long)Hc * K2;
    const __bf16* pWl_z = pWl_r + (long)Hc * K2;
    const __bf16* pWh_n = pWh_r + (long)2 * Hc * K2;
    const __bf16* pWl_n = pWl_r + (long)2 * Hc * K2;

    v8f acc_r = {}, acc_z = {}, acc_in = {}, acc_hn = {};
    const v16bf vzero = {};

    // ---- Phase 1: K in [0,1024) — x contribution ------------------------
#pragma unroll 2
    for (int kk = 0; kk < 32; ++kk) {
        const int kb = kk * 32;
        v16bf ah = *(const v16bf*)(pxh + kb);
        v16bf al = *(const v16bf*)(pxl + kb);
        v16bf bh_r = *(const v16bf*)(pWh_r + kb), bl_r = *(const v16bf*)(pWl_r + kb);
        v16bf bh_z = *(const v16bf*)(pWh_z + kb), bl_z = *(const v16bf*)(pWl_z + kb);
        v16bf bh_n = *(const v16bf*)(pWh_n + kb), bl_n = *(const v16bf*)(pWl_n + kb);
        acc_r  = mm3(ah, al, bh_r, bl_r, acc_r);
        acc_z  = mm3(ah, al, bh_z, bl_z, acc_z);
        acc_in = mm3(ah, al, bh_n, bl_n, acc_in);  // n-gate x part kept separate
    }
    // ---- Phase 2: K in [1024,2048) — masked-h contribution --------------
#pragma unroll 2
    for (int kk = 0; kk < 32; ++kk) {
        const int kb = kk * 32;
        const int kw = kb + 1024;
        v16bf ah = *(const v16bf*)(phh + kb);
        v16bf al = *(const v16bf*)(phl + kb);
        ah = init ? vzero : ah;   // h := h * mask (branchless select)
        al = init ? vzero : al;
        v16bf bh_r = *(const v16bf*)(pWh_r + kw), bl_r = *(const v16bf*)(pWl_r + kw);
        v16bf bh_z = *(const v16bf*)(pWh_z + kw), bl_z = *(const v16bf*)(pWl_z + kw);
        v16bf bh_n = *(const v16bf*)(pWh_n + kw), bl_n = *(const v16bf*)(pWl_n + kw);
        acc_r  = mm3(ah, al, bh_r, bl_r, acc_r);
        acc_z  = mm3(ah, al, bh_z, bl_z, acc_z);
        acc_hn = mm3(ah, al, bh_n, bl_n, acc_hn);  // n-gate h part kept separate
    }

    // ---- Gate nonlinearities + recurrence, in-register ------------------
    // C/D layout: VGPR r -> M = r (lanes 0-15) or r+8 (lanes 16-31); N = lane&15.
    const int   col = nh * 16 + lm;
    const int   psw = swz(col);               // swizzled position for hbuf write
    const float br  = b_ih[col] + b_hh[col];
    const float bz  = b_ih[Hc + col] + b_hh[Hc + col];
    const float bin = b_ih[2 * Hc + col];
    const float bhn = b_hh[2 * Hc + col];
#pragma unroll
    for (int r = 0; r < 8; ++r) {
        const int   row = m0 + (hs ? r + 8 : r);
        const float m   = is_init[(long)row * Tc + t] ? 0.0f : 1.0f;
        const float hm  = hprev[(long)row * hstride + col] * m;
        const float rg  = sigmoidf_(acc_r[r] + br);
        const float zg  = sigmoidf_(acc_z[r] + bz);
        const float ng  = tanhf(acc_in[r] + bin + rg * (acc_hn[r] + bhn));
        const float hn  = (1.0f - zg) * ng + zg * hm;
        hseq[((long)row * Tc + t) * Hc + col] = hn;
        const __bf16 hh = (__bf16)hn;                 // bf16 hi/lo for next step
        hbw_h[(long)row * Hc + psw] = hh;
        hbw_l[(long)row * Hc + psw] = (__bf16)(hn - (float)hh);
    }
}

// ---------------------------------------------------------------------------
// Kernel 5: Y = LayerNorm(Hseq)*g + b + x*sigmoid(res_gate). Block per row.
// ---------------------------------------------------------------------------
__global__ __launch_bounds__(256) void ln_res(const float* __restrict__ hseq,
                                              const float* __restrict__ x,
                                              const float* __restrict__ ln_g,
                                              const float* __restrict__ ln_b,
                                              const float* __restrict__ res_gate,
                                              float* __restrict__ Y) {
    const long row = blockIdx.x;               // b*T + t
    const float* hv = hseq + row * Hc;
    const float* xv = x + row * Dc;
    __shared__ float s1[256], s2[256];
    const int tid = threadIdx.x;
    float a = 0.0f, b = 0.0f;
    for (int i = tid; i < Hc; i += 256) {
        float v = hv[i];
        a += v;
        b += v * v;
    }
    s1[tid] = a; s2[tid] = b;
    __syncthreads();
    for (int off = 128; off > 0; off >>= 1) {
        if (tid < off) { s1[tid] += s1[tid + off]; s2[tid] += s2[tid + off]; }
        __syncthreads();
    }
    const float mu   = s1[0] * (1.0f / Hc);
    const float var  = s2[0] * (1.0f / Hc) - mu * mu;
    const float rstd = rsqrtf(var + 1e-5f);
    for (int i = tid; i < Hc; i += 256) {
        float v = hv[i];
        Y[row * Hc + i] = (v - mu) * rstd * ln_g[i] + ln_b[i] + xv[i] * sigmoidf_(res_gate[i]);
    }
}

// ---------------------------------------------------------------------------
extern "C" void kernel_launch(void* const* d_in, const int* in_sizes, int n_in,
                              void* d_out, int out_size, void* d_ws, size_t ws_size,
                              hipStream_t stream) {
    const float*         x        = (const float*)d_in[0];
    const float*         hx       = (const float*)d_in[1];
    const unsigned char* is_init  = (const unsigned char*)d_in[2]; // jax bool = 1 byte
    const float*         W_ih     = (const float*)d_in[3];
    const float*         W_hh     = (const float*)d_in[4];
    const float*         b_ih     = (const float*)d_in[5];
    const float*         b_hh     = (const float*)d_in[6];
    const float*         ln_g     = (const float*)d_in[7];
    const float*         ln_b     = (const float*)d_in[8];
    const float*         res_gate = (const float*)d_in[9];

    float* Y    = (float*)d_out;                       // (B,T,H)
    float* Hseq = Y + (long)Bc * Tc * Hc;              // (B,T,H)

    // workspace layout
    __bf16* Wh  = (__bf16*)d_ws;                       // G*K2
    __bf16* Wl  = Wh + (long)Gc * K2;                  // G*K2
    __bf16* Xh  = Wl + (long)Gc * K2;                  // B*T*D (swizzled)
    __bf16* Xl  = Xh + (long)Bc * Tc * Dc;             // B*T*D (swizzled)
    __bf16* Hb0h = Xl + (long)Bc * Tc * Dc;            // ping-pong h buffers
    __bf16* Hb0l = Hb0h + (long)Bc * Hc;
    __bf16* Hb1h = Hb0l + (long)Bc * Hc;
    __bf16* Hb1l = Hb1h + (long)Bc * Hc;

    // 1) conversions (parallel, every call: deterministic)
    {
        long n = (long)Gc * K2;
        conv_w<<<(unsigned)((n + 255) / 256), 256, 0, stream>>>(W_ih, W_hh, Wh, Wl);
        long nx = (long)Bc * Tc * Dc;
        conv_x<<<(unsigned)((nx + 255) / 256), 256, 0, stream>>>(x, Xh, Xl);
        long nh = (long)Bc * Hc;
        conv_h0<<<(unsigned)((nh + 255) / 256), 256, 0, stream>>>(hx, Hb0h, Hb0l);
    }

    // 2) the sequential scan: 128 stream-ordered step launches
    for (int t = 0; t < Tc; ++t) {
        const __bf16* rh = (t & 1) ? Hb1h : Hb0h;
        const __bf16* rl = (t & 1) ? Hb1l : Hb0l;
        __bf16*       wh = (t & 1) ? Hb0h : Hb1h;
        __bf16*       wl = (t & 1) ? Hb0l : Hb1l;
        const float* hprev   = (t == 0) ? hx : (Hseq + (long)(t - 1) * Hc);
        const long   hstride = (t == 0) ? (long)Hc : (long)Tc * Hc;
        gru_step<<<dim3(Hc / 16, Bc / 16), 32, 0, stream>>>(
            Xh, Xl, rh, rl, wh, wl, hprev, hstride, is_init, Wh, Wl, b_ih, b_hh, Hseq, t);
    }

    // 3) layernorm + gated residual
    ln_res<<<Bc * Tc, 256, 0, stream>>>(Hseq, x, ln_g, ln_b, res_gate, Y);
}
